// ManualLSTM_53996328845818
// MI455X (gfx1250) — compile-verified, matrix operand order checked by reading, not run on hardware
//
#include <hip/hip_runtime.h>

// ---------------------------------------------------------------------------
// LSTM  B=128, T=1024, D=256, HS=512  (gates G = 4*HS = 2048)
// Phase 1: xW[B*T, G] = x @ W + bias   (bf16 WMMA, f32 accum)
// Phase 2: per-batch-slice recurrence  h,c update with h@U via WMMA
// Workspace layout (d_ws), ~1.07 GiB required:
//   [0]                xbf  : B*T*D bf16           (64 MiB)
//   [+64MiB]           Wp   : packed W fragments    (1 MiB)
//   [+65MiB]           Up   : packed U fragments    (2 MiB)
//   [+67MiB]           xW   : B*T*G f32             (1 GiB)
// ---------------------------------------------------------------------------

#define BB   128
#define TT   1024
#define DD   256
#define HHS  512
#define GG   2048           // 4*HS
#define BT   (BB*TT)        // 131072

typedef __bf16 bfx8  __attribute__((ext_vector_type(8)));
typedef __bf16 bfx16 __attribute__((ext_vector_type(16)));
typedef float  fx8   __attribute__((ext_vector_type(8)));

#define WMMA_BF16(a, b, c) \
  __builtin_amdgcn_wmma_f32_16x16x32_bf16(false, (a), false, (b), (short)0, (c), false, false)

__device__ __forceinline__ __bf16 f2bf(float f) {
  unsigned u = __builtin_bit_cast(unsigned, f);
  unsigned r = u + 0x7FFFu + ((u >> 16) & 1u);   // round-to-nearest-even
  unsigned short s = (unsigned short)(r >> 16);
  return __builtin_bit_cast(__bf16, s);
}

__device__ __forceinline__ float sigmoidf_(float x) {
  return 1.0f / (1.0f + __expf(-x));
}

// --------------------------- convert x -> bf16 -----------------------------
__global__ __launch_bounds__(256) void cvt_bf16_kernel(
    const float* __restrict__ src, __bf16* __restrict__ dst, int n) {
  int i = blockIdx.x * 256 + threadIdx.x;
  if (i < n) dst[i] = f2bf(src[i]);
}

// ----------- pack row-major [K][N] f32 into WMMA B-fragment layout ---------
// dst index = ((kc*(N/16) + nt)*32 + lane)*16 + e
// lane: n = nt*16 + (lane&15); koff = (lane>>4)*8
// e<8 : k = kc*32 + koff + e ; e>=8 : k = kc*32 + 16 + koff + (e-8)
__global__ __launch_bounds__(256) void pack_b_kernel(
    const float* __restrict__ src, __bf16* __restrict__ dst, int K, int N) {
  int tid = blockIdx.x * 256 + threadIdx.x;
  if (tid >= K * N) return;
  int e    = tid & 15;
  int lane = (tid >> 4) & 31;
  int rest = tid >> 9;
  int ntiles = N >> 4;
  int nt = rest % ntiles;
  int kc = rest / ntiles;
  int n  = nt * 16 + (lane & 15);
  int koff = (lane >> 4) << 3;
  int k = kc * 32 + ((e < 8) ? (koff + e) : (16 + koff + (e - 8)));
  dst[tid] = f2bf(src[(size_t)k * N + n]);
}

// ------------------------- GEMM1: xW = x@W + bias --------------------------
// grid = (BT/128, G/64), block = 256 (8 waves). Wave computes 16(M) x 64(N).
__global__ __launch_bounds__(256) void gemm_xw_kernel(
    const __bf16* __restrict__ A,    // [BT][D] row-major bf16
    const __bf16* __restrict__ Bp,   // packed W fragments
    const float*  __restrict__ bias, // [G]
    float* __restrict__ C) {         // [BT][G] f32
  const int lane = threadIdx.x & 31;
  const int w    = threadIdx.x >> 5;
  const int ln   = lane & 15;
  const int lh   = lane >> 4;
  const int m0   = blockIdx.x * 128 + w * 16;
  const int n0   = blockIdx.y * 64;

  fx8 acc[4];
#pragma unroll
  for (int j = 0; j < 4; ++j) acc[j] = (fx8){};

  const __bf16* arow = A + (size_t)(m0 + ln) * DD + (lh << 3);
  const int nt0 = n0 >> 4;

#pragma unroll
  for (int kc = 0; kc < DD / 32; ++kc) {
    const __bf16* ap = arow + kc * 32;
    bfx8 lo = *(const bfx8*)ap;
    bfx8 hi = *(const bfx8*)(ap + 16);
    bfx16 a = __builtin_shufflevector(lo, hi, 0,1,2,3,4,5,6,7,8,9,10,11,12,13,14,15);
    const __bf16* bp = Bp + ((size_t)(kc * (GG / 16) + nt0) * 32 + lane) * 16;
#pragma unroll
    for (int j = 0; j < 4; ++j) {
      bfx16 b = *(const bfx16*)(bp + (size_t)j * 512);  // 32*16 elems per n-tile
      acc[j] = WMMA_BF16(a, b, acc[j]);
    }
  }

#pragma unroll
  for (int j = 0; j < 4; ++j) {
    int n = n0 + j * 16 + ln;
    float bv = bias[n];
#pragma unroll
    for (int r = 0; r < 8; ++r) {
      int m = m0 + r + (lh << 3);
      C[(size_t)m * GG + n] = acc[j][r] + bv;
    }
  }
}

// --------------------------- LSTM recurrence -------------------------------
// grid = 8 blocks (16 batch rows each), block = 512 threads (16 waves).
// Per step: gates[16,2048] = xW_t + h@U (WMMA), activate, then cell update.
#define HROW 520    // 512 + 8 bf16 pad (kills LDS bank conflicts on A reads)
#define GROW 2064   // 2048 + 16 f32 pad

__global__ __launch_bounds__(512) void lstm_rec_kernel(
    const __bf16* __restrict__ Up,   // packed U fragments
    const float*  __restrict__ xW,   // [BT][G]
    float* __restrict__ hs,          // [B][T][HS]
    float* __restrict__ hT,          // [B][HS]
    float* __restrict__ cT) {        // [B][HS]
  __shared__ float  gsh[16 * GROW];  // activated gates, f32
  __shared__ __bf16 hbf[16 * HROW];  // h slice, bf16 row-major (WMMA A source)

  const int tid   = threadIdx.x;
  const int lane  = tid & 31;
  const int w     = tid >> 5;
  const int ln    = lane & 15;
  const int lh    = lane >> 4;
  const int bbase = blockIdx.x * 16;
  const int gsel  = w >> 2;          // 0:i 1:f 2:g 3:o (uniform per wave)
  const int urow  = tid >> 5;        // update-phase row
  const int ucb   = (tid & 31) * 16; // update-phase col base

  for (int i = tid; i < 16 * HROW; i += 512) hbf[i] = f2bf(0.0f);
  __syncthreads();

  float creg[16];
#pragma unroll
  for (int j = 0; j < 16; ++j) creg[j] = 0.0f;

  for (int t = 0; t < TT; ++t) {
    // ---- gate phase: wave w computes n-tiles [w*8, w*8+8) -----------------
    for (int j = 0; j < 8; ++j) {
      const int nt = w * 8 + j;
      fx8 acc = (fx8){};
#pragma unroll
      for (int kc = 0; kc < HHS / 32; ++kc) {
        const __bf16* ap = hbf + ln * HROW + kc * 32 + (lh << 3);
        bfx8 lo = *(const bfx8*)ap;
        bfx8 hi = *(const bfx8*)(ap + 16);
        bfx16 a = __builtin_shufflevector(lo, hi, 0,1,2,3,4,5,6,7,8,9,10,11,12,13,14,15);
        bfx16 b = *(const bfx16*)(Up + ((size_t)(kc * (GG / 16) + nt) * 32 + lane) * 16);
        acc = WMMA_BF16(a, b, acc);
      }
      const int n = nt * 16 + ln;
#pragma unroll
      for (int r = 0; r < 8; ++r) {
        const int m = r + (lh << 3);
        float v = acc[r] + xW[((size_t)(bbase + m) * TT + t) * GG + n];
        float act = (gsel == 2) ? tanhf(v) : sigmoidf_(v);
        gsh[m * GROW + n] = act;
      }
    }
    __syncthreads();

    // ---- update phase: thread owns (urow, cols ucb..ucb+15) --------------
#pragma unroll
    for (int j = 0; j < 16; ++j) {
      const int col = ucb + j;
      float i_ = gsh[urow * GROW + col];
      float f_ = gsh[urow * GROW + 512 + col];
      float g_ = gsh[urow * GROW + 1024 + col];
      float o_ = gsh[urow * GROW + 1536 + col];
      float c  = f_ * creg[j] + i_ * g_;
      creg[j]  = c;
      float h  = o_ * tanhf(c);
      hs[((size_t)(bbase + urow) * TT + t) * HHS + col] = h;
      hbf[urow * HROW + col] = f2bf(h);
      if (t == TT - 1) {
        hT[(size_t)(bbase + urow) * HHS + col] = h;
        cT[(size_t)(bbase + urow) * HHS + col] = c;
      }
    }
    __syncthreads();
  }
}

// ---------------------------------------------------------------------------
extern "C" void kernel_launch(void* const* d_in, const int* in_sizes, int n_in,
                              void* d_out, int out_size, void* d_ws, size_t ws_size,
                              hipStream_t stream) {
  const float* x    = (const float*)d_in[0];   // [B,T,D]
  const float* W    = (const float*)d_in[1];   // [D,G]
  const float* U    = (const float*)d_in[2];   // [HS,G]
  const float* bias = (const float*)d_in[3];   // [G]

  char* ws = (char*)d_ws;
  const size_t xbf_bytes = (size_t)BT * DD * 2;        // 64 MiB
  const size_t wp_bytes  = (size_t)DD * GG * 2;        //  1 MiB
  const size_t up_bytes  = (size_t)HHS * GG * 2;       //  2 MiB
  __bf16* xbf = (__bf16*)ws;
  __bf16* Wp  = (__bf16*)(ws + xbf_bytes);
  __bf16* Up  = (__bf16*)(ws + xbf_bytes + wp_bytes);
  float*  xW  = (float*)(ws + xbf_bytes + wp_bytes + up_bytes);

  float* hs = (float*)d_out;                           // [B,T,HS]
  float* hT = hs + (size_t)BB * TT * HHS;              // [B,HS]
  float* cT = hT + (size_t)BB * HHS;                   // [B,HS]

  // 1) x -> bf16
  {
    int n = BT * DD;
    cvt_bf16_kernel<<<(n + 255) / 256, 256, 0, stream>>>(x, xbf, n);
  }
  // 2) pack W, U into WMMA B-fragment layout
  pack_b_kernel<<<(DD * GG + 255) / 256, 256, 0, stream>>>(W, Wp, DD, GG);
  pack_b_kernel<<<(HHS * GG + 255) / 256, 256, 0, stream>>>(U, Up, HHS, GG);
  // 3) xW = x@W + bias
  gemm_xw_kernel<<<dim3(BT / 128, GG / 64), 256, 0, stream>>>(xbf, Wp, bias, xW);
  // 4) recurrence
  lstm_rec_kernel<<<BB / 16, 512, 0, stream>>>(Up, xW, hs, hT, cT);
}
